// MHMLP_89893665505618
// MI455X (gfx1250) — compile-verified, matrix operand order checked by reading, not run on hardware
//
#include <hip/hip_runtime.h>

// Problem constants (match reference setup_inputs)
constexpr int B    = 2048;
constexpr int D    = 512;
constexpr int H    = 64;
constexpr int HID  = 2048;

constexpr int CHUNK  = 128;          // HID columns handled per workgroup
constexpr int NCHUNK = HID / CHUNK;  // 16 chunks
constexpr int WAVES  = 8;            // 256 threads / wave32
constexpr int NPW    = CHUNK / WAVES;// 16 columns per wave (one WMMA N tile)
constexpr int KSTEPS = D / 32;       // 16 k-steps of K=32
constexpr int MTILE  = 32;           // batch rows per iteration (2 WMMA row blocks)
constexpr int MTILES = B / MTILE;    // 64 iterations

typedef _Float16 half_t;
typedef __attribute__((ext_vector_type(16))) _Float16 v16h;
typedef __attribute__((ext_vector_type(8)))  _Float16 v8h;
typedef __attribute__((ext_vector_type(8)))  float    v8f;
typedef __attribute__((ext_vector_type(4)))  float    v4f;

// ISA 7.12.2: 16-bit A/B fragment k-mapping for 16x16x32.
// lane<16 : j=0..7 -> k=j,     j=8..15 -> k=16+(j-8)
// lane>=16: j=0..7 -> k=8+j,   j=8..15 -> k=24+(j-8)
__device__ __forceinline__ int kmap(int lane, int j) {
    int lo = (lane < 16) ? 0 : 8;
    int hi = (lane < 16) ? 16 : 24;
    return (j < 8) ? (lo + j) : (hi + (j - 8));
}

__global__ __launch_bounds__(256)
void mhmlp_stage1(const float* __restrict__ x,
                  const float* __restrict__ W1,
                  const float* __restrict__ b1,
                  const float* __restrict__ W2,
                  float* __restrict__ ws) {
    const int chunk = blockIdx.x;
    const int head  = blockIdx.y;
    const int tid   = threadIdx.x;
    const int wave  = tid >> 5;
    const int lane  = tid & 31;
    const int lane16 = lane & 15;
    const bool hiHalf = (lane >= 16);

    // Two row-blocks of A fragments, pre-swizzled so each lane's frag is one 32B read.
    __shared__ __align__(32) half_t lds_x[2][KSTEPS][32][16];   // 32 KB
    __shared__ float lds_part[2][WAVES][MTILE];                 // double-buffered partials

    // ---- One-time: this wave's 16 W1 columns as f16 B-fragments held in VGPRs
    // for the entire kernel => W1 read from HBM exactly once overall.
    // Non-temporal: W1 is a 256MB one-shot stream; keep it out of L2 so the
    // x matrix (4MB, re-read by all workgroups) stays resident.
    const int ncol = chunk * CHUNK + wave * NPW + lane16;
    const float* W1h = W1 + (size_t)head * D * HID;
    v16h Bfrag[KSTEPS];
#pragma unroll
    for (int kk = 0; kk < KSTEPS; ++kk) {
#pragma unroll
        for (int j = 0; j < 16; ++j) {
            int k = kk * 32 + kmap(lane, j);
            ((half_t*)&Bfrag[kk])[j] =
                (half_t)__builtin_nontemporal_load(&W1h[(size_t)k * HID + ncol]);
        }
    }
    const float b1v = b1[head * HID + ncol];
    const float w2v = W2[head * HID + ncol];

    for (int mt = 0; mt < MTILES; ++mt) {
        const int m0 = mt * MTILE;
        __syncthreads();  // A: prior compute done with lds_x / lds_part readable

        // Deferred: emit previous tile's per-chunk partials (overlaps with staging).
        if (mt > 0 && tid < MTILE) {
            const int prev = mt - 1;
            float s = 0.0f;
#pragma unroll
            for (int w = 0; w < WAVES; ++w) s += lds_part[prev & 1][w][tid];
            ws[((size_t)chunk * B + (prev * MTILE + tid)) * H + head] = s;
        }

        // Prefetch next tile's rows (global_prefetch_b8).
        if (mt + 1 < MTILES) {
            __builtin_prefetch(x + (size_t)(m0 + MTILE + (tid & 31)) * D + (tid >> 5) * 64, 0, 3);
        }

        // Stage x tile (32 rows x 512 cols) fp32 -> f16 into fragment layout.
        // Unit = 8 consecutive k's for one (rowblock, kstep, laneslot, half):
        // 2 x global_load_b128 -> packed cvt -> 1 x 16B ds_store.
#pragma unroll
        for (int i = 0; i < 8; ++i) {
            int u  = tid + i * 256;           // 2048 units total
            int jh = u & 1;
            int ls = (u >> 1) & 31;
            int kk = (u >> 6) & 15;
            int rb = u >> 10;
            int lo = (ls < 16) ? 0 : 8;
            int hi = (ls < 16) ? 16 : 24;
            int kcol = kk * 32 + (jh ? hi : lo);
            int row  = m0 + rb * 16 + (ls & 15);
            const v4f* gp = (const v4f*)(x + (size_t)row * D + kcol);
            v4f f0 = gp[0];
            v4f f1 = gp[1];
            v8h hv;
#pragma unroll
            for (int t = 0; t < 4; ++t) hv[t]     = (half_t)f0[t];
#pragma unroll
            for (int t = 0; t < 4; ++t) hv[4 + t] = (half_t)f1[t];
            *(v8h*)&lds_x[rb][kk][ls][jh * 8] = hv;
        }
        __syncthreads();  // B: tile staged

        // Two 16x16 output tiles per wave. 2-deep rotating A-fragment pipeline;
        // sched_group_barriers pin the schedule to [2 WMMA][4 DS-read] per
        // k-step so refill loads get ~2 WMMA-pairs of latency slack instead of
        // being sunk to just before their consumer (s_wait_dscnt 0x0).
        v8f acc0 = {};
        v8f acc1 = {};
        v16h af[2][2];
        af[0][0] = *(const v16h*)&lds_x[0][0][lane][0];
        af[0][1] = *(const v16h*)&lds_x[1][0][lane][0];
        af[1][0] = *(const v16h*)&lds_x[0][1][lane][0];
        af[1][1] = *(const v16h*)&lds_x[1][1][lane][0];
        __builtin_amdgcn_sched_barrier(0);   // fence init loads out of the pattern
#pragma unroll
        for (int kk = 0; kk < KSTEPS; ++kk) {
            const int par = kk & 1;
            acc0 = __builtin_amdgcn_wmma_f32_16x16x32_f16(
                false, af[par][0], false, Bfrag[kk], (short)0, acc0, false, false);
            acc1 = __builtin_amdgcn_wmma_f32_16x16x32_f16(
                false, af[par][1], false, Bfrag[kk], (short)0, acc1, false, false);
            if (kk + 2 < KSTEPS) {
                af[par][0] = *(const v16h*)&lds_x[0][kk + 2][lane][0];
                af[par][1] = *(const v16h*)&lds_x[1][kk + 2][lane][0];
            }
            __builtin_amdgcn_sched_group_barrier(0x008, 2, 0);   // 2 WMMA
            if (kk + 2 < KSTEPS)
                __builtin_amdgcn_sched_group_barrier(0x100, 4, 0); // 4 DS reads
        }
        __builtin_amdgcn_sched_barrier(0);   // end of pinned region

        // Fused stage 2: h = leakyrelu(acc + b1); partial = h * W2[n];
        // reduce over the 16 columns (lane dim) per half-wave.
        float p0[8], p1[8];
#pragma unroll
        for (int v = 0; v < 8; ++v) {
            float t0 = acc0[v] + b1v;
            t0 = (t0 >= 0.0f) ? t0 : 0.01f * t0;
            p0[v] = t0 * w2v;
            float t1 = acc1[v] + b1v;
            t1 = (t1 >= 0.0f) ? t1 : 0.01f * t1;
            p1[v] = t1 * w2v;
        }
#pragma unroll
        for (int off = 1; off <= 8; off <<= 1) {
#pragma unroll
            for (int v = 0; v < 8; ++v) {
                p0[v] += __shfl_xor(p0[v], off, 32);
                p1[v] += __shfl_xor(p1[v], off, 32);
            }
        }
        if (lane16 == 0) {
            int base = hiHalf ? 8 : 0;
#pragma unroll
            for (int v = 0; v < 8; ++v) {
                lds_part[mt & 1][wave][base + v]      = p0[v];
                lds_part[mt & 1][wave][16 + base + v] = p1[v];
            }
        }
    }

    // Epilogue: flush the last tile's partials.
    __syncthreads();
    if (tid < MTILE) {
        const int prev = MTILES - 1;
        float s = 0.0f;
#pragma unroll
        for (int w = 0; w < WAVES; ++w) s += lds_part[prev & 1][w][tid];
        ws[((size_t)chunk * B + (prev * MTILE + tid)) * H + head] = s;
    }
}

__global__ __launch_bounds__(256)
void mhmlp_stage2(const float* __restrict__ ws,
                  const float* __restrict__ b2,
                  float* __restrict__ out) {
    int i = blockIdx.x * blockDim.x + threadIdx.x;   // i = b*H + h
    if (i < B * H) {
        float s = b2[i & (H - 1)];
#pragma unroll
        for (int c = 0; c < NCHUNK; ++c) s += ws[(size_t)c * B * H + i];
        out[i] = (s >= 0.0f) ? s : 0.01f * s;
    }
}

extern "C" void kernel_launch(void* const* d_in, const int* in_sizes, int n_in,
                              void* d_out, int out_size, void* d_ws, size_t ws_size,
                              hipStream_t stream) {
    const float* x  = (const float*)d_in[0];   // [B, D]
    const float* W1 = (const float*)d_in[1];   // [H, D, HID]
    const float* b1 = (const float*)d_in[2];   // [H, HID]
    const float* W2 = (const float*)d_in[3];   // [H, HID]
    const float* b2 = (const float*)d_in[4];   // [H]
    float* out = (float*)d_out;                // [B, H]
    float* ws  = (float*)d_ws;                 // [NCHUNK, B, H] partials (8 MB)

    dim3 grid1(NCHUNK, H);
    mhmlp_stage1<<<grid1, 256, 0, stream>>>(x, W1, b1, W2, ws);

    int total = B * H;
    mhmlp_stage2<<<(total + 255) / 256, 256, 0, stream>>>(ws, b2, out);
}
